// GraphSAGE_90606630076836
// MI455X (gfx1250) — matrix operand order repaired and checked by reading, not compiled, not run on hardware
//
#include <hip/hip_runtime.h>

// ---------------------------------------------------------------------------
// MI455X (gfx1250) GraphSAGE forward.
//   - bf16 features + v_wmma_f32_16x16x32_bf16 for all dense GEMMs
//   - weights staged into LDS via global_load_async_to_lds_b128 (ASYNCcnt)
//   - f32 atomic scatter-add for edge aggregation (agg buffers fit in 192MB L2)
//   - wave32 shuffle reductions for L2-norm and edge dot product
// ---------------------------------------------------------------------------

typedef __bf16 bf16;
typedef bf16  v16bf __attribute__((ext_vector_type(16)));
typedef bf16  v8bf  __attribute__((ext_vector_type(8)));
typedef float v8f   __attribute__((ext_vector_type(8)));

static constexpr int HD = 128;   // hidden dim
static constexpr int OD = 64;    // output dim
static constexpr int FUD = 32;   // user raw feature dim
static constexpr int FAD = 64;   // app raw feature dim
static constexpr int TPB = 256;  // threads per block (8 waves, wave32)

// ------------------------------- utility -----------------------------------

__global__ void zero_f32_k(float* __restrict__ p, long n) {
  long i = (long)blockIdx.x * blockDim.x + threadIdx.x;
  long stride = (long)gridDim.x * blockDim.x;
  for (; i < n; i += stride) p[i] = 0.0f;
}

__global__ void cvt_bf16_k(const float* __restrict__ w, bf16* __restrict__ o, int n) {
  int i = blockIdx.x * blockDim.x + threadIdx.x;
  if (i < n) o[i] = (bf16)w[i];
}

// ------------------------------- degrees -----------------------------------

__global__ void deg_k(const int* __restrict__ es, const int* __restrict__ ed,
                      float* __restrict__ du, float* __restrict__ da, int E) {
  int i = blockIdx.x * blockDim.x + threadIdx.x;
  if (i < E) {
    atomicAdd(du + es[i], 1.0f);
    atomicAdd(da + ed[i], 1.0f);
  }
}

// ------------------------- edge scatter (both dirs) -------------------------
// agg_u[src] += fa[dst]   (app features -> user agg)
// agg_a[dst] += fu[src]   (user features -> app agg)
// one thread = one edge x 4 channels
__global__ void scatter_k(const int* __restrict__ es, const int* __restrict__ ed,
                          const bf16* __restrict__ fu, const bf16* __restrict__ fa,
                          float* __restrict__ agg_u, float* __restrict__ agg_a, int E) {
  long tid = (long)blockIdx.x * blockDim.x + threadIdx.x;
  long e = tid >> 5;
  if (e >= E) return;
  int c = (int)(tid & 31) * 4;
  int s = es[e], d = ed[e];
  const bf16* pa = fa + (long)d * HD + c;
  float* qu = agg_u + (long)s * HD + c;
#pragma unroll
  for (int i = 0; i < 4; i++) atomicAdd(qu + i, (float)pa[i]);
  const bf16* pu = fu + (long)s * HD + c;
  float* qa = agg_a + (long)d * HD + c;
#pragma unroll
  for (int i = 0; i < 4; i++) atomicAdd(qa + i, (float)pu[i]);
}

// --------------------- input encoder: emb[nid] + x @ W^T + b ----------------
// One wave computes a 16(M) x 128(N) strip: 8 accumulator tiles, K = FK.
// A layout (16-bit, 16x32): lane = {h=lane/16, m=lane%16}; elems 0..15 -> K=k0+16h..
// contiguously (same k order as B).
// B layout (32x16): lane half h holds K=k0+16h..+15 contiguously -> read 16
// contiguous bf16 of W row n_g (since B[k][n] = W[n][k]).
template <int FK>
__global__ __launch_bounds__(TPB) void encode_k(
    const int* __restrict__ nid, const float* __restrict__ x,
    const float* __restrict__ emb, const bf16* __restrict__ Wb,
    const float* __restrict__ bias, bf16* __restrict__ out, int M) {
  int wave = (int)(((long)blockIdx.x * blockDim.x + threadIdx.x) >> 5);
  int strip = wave;
  if (strip * 16 >= M) return;            // wave-uniform exit (EXEC stays all-1 for WMMA)
  int lane = threadIdx.x & 31;
  int h = lane >> 4, ln = lane & 15;
  long arow = (long)strip * 16 + ln;

  v8f c[8] = {};
#pragma unroll
  for (int k0 = 0; k0 < FK; k0 += 32) {
    const float* p0 = x + arow * FK + k0 + 16 * h;
    const float* p1 = p0 + 8;
    v16bf a;
#pragma unroll
    for (int i = 0; i < 8; i++) { a[i] = (bf16)p0[i]; a[8 + i] = (bf16)p1[i]; }
#pragma unroll
    for (int t = 0; t < 8; t++) {
      v16bf b = *(const v16bf*)(Wb + (t * 16 + ln) * FK + k0 + 16 * h);
      c[t] = __builtin_amdgcn_wmma_f32_16x16x32_bf16(false, a, false, b,
                                                     (short)0, c[t], false, false);
    }
  }
  // D layout: VGPR r holds row m = r + 8h, column = t*16 + ln.
  long base[8];
#pragma unroll
  for (int r = 0; r < 8; r++)
    base[r] = (long)nid[strip * 16 + r + 8 * h] * HD;
#pragma unroll
  for (int t = 0; t < 8; t++) {
    float bv = bias[t * 16 + ln];
#pragma unroll
    for (int r = 0; r < 8; r++) {
      float v = c[t][r] + bv + emb[base[r] + t * 16 + ln];
      out[((long)strip * 16 + r + 8 * h) * HD + t * 16 + ln] = (bf16)v;
    }
  }
}

// --------------------------- SAGE conv layer --------------------------------
// out = (agg/deg) @ Wl^T + bias + self @ Wr^T ; optional L2-normalize + relu.
// One wave -> 16 x N strip, K = 128.  Both weight matrices are staged into LDS
// once per block with async-to-LDS loads (ASYNCcnt), then every wave reads its
// B fragments from LDS, keeping the WGP$ free for the streaming A operands.
template <int N, bool NORMRELU, bool OUTBF16>
__global__ __launch_bounds__(TPB) void sage_k(
    const float* __restrict__ agg, const float* __restrict__ deg,
    const bf16* __restrict__ sf,
    const bf16* __restrict__ Wl, const bf16* __restrict__ Wr,
    const float* __restrict__ bias, void* __restrict__ outv, int M) {
  constexpr int NT = N / 16;
  __shared__ __align__(64) bf16 sWl[N * HD];
  __shared__ __align__(64) bf16 sWr[N * HD];

  // ---- async stage weights into LDS (all waves participate, then barrier) ----
  {
    constexpr int BYTES = N * HD * 2;       // per matrix
    constexpr int ITER = BYTES / (TPB * 16);
    uint32_t t = threadIdx.x;
    uint32_t l0 = (uint32_t)(uintptr_t)&sWl[0];   // flat->LDS offset (addr[31:0])
    uint32_t l1 = (uint32_t)(uintptr_t)&sWr[0];
    const char* g0 = (const char*)Wl;
    const char* g1 = (const char*)Wr;
#pragma unroll
    for (int i = 0; i < ITER; i++) {
      uint32_t boff = (i * TPB + t) * 16;
      uint32_t la0 = l0 + boff, la1 = l1 + boff;
      uint64_t ga0 = (uint64_t)(uintptr_t)(g0 + boff);
      uint64_t ga1 = (uint64_t)(uintptr_t)(g1 + boff);
      asm volatile("global_load_async_to_lds_b128 %0, %1, off"
                   :: "v"(la0), "v"(ga0) : "memory");
      asm volatile("global_load_async_to_lds_b128 %0, %1, off"
                   :: "v"(la1), "v"(ga1) : "memory");
    }
    asm volatile("s_wait_asynccnt 0" ::: "memory");
  }
  __syncthreads();

  int wave = (int)(((long)blockIdx.x * blockDim.x + threadIdx.x) >> 5);
  int strip = wave;
  if (strip * 16 >= M) return;            // after barrier: wave-uniform exit
  int lane = threadIdx.x & 31;
  int h = lane >> 4, ln = lane & 15;
  long arow = (long)strip * 16 + ln;
  float rdeg = 1.0f / fmaxf(deg[arow], 1.0f);

  v8f c[NT] = {};
  // neighbor-mean term
#pragma unroll
  for (int k0 = 0; k0 < HD; k0 += 32) {
    const float* p0 = agg + arow * HD + k0 + 16 * h;
    v16bf a;
#pragma unroll
    for (int i = 0; i < 8; i++) {
      a[i]     = (bf16)(p0[i] * rdeg);
      a[8 + i] = (bf16)(p0[8 + i] * rdeg);
    }
#pragma unroll
    for (int t = 0; t < NT; t++) {
      v16bf b = *(const v16bf*)(sWl + (t * 16 + ln) * HD + k0 + 16 * h);
      c[t] = __builtin_amdgcn_wmma_f32_16x16x32_bf16(false, a, false, b,
                                                     (short)0, c[t], false, false);
    }
  }
  // self term
#pragma unroll
  for (int k0 = 0; k0 < HD; k0 += 32) {
    v8bf lo = *(const v8bf*)(sf + arow * HD + k0 + 16 * h);
    v8bf hi = *(const v8bf*)(sf + arow * HD + k0 + 8 + 16 * h);
    v16bf a;
#pragma unroll
    for (int i = 0; i < 8; i++) { a[i] = lo[i]; a[8 + i] = hi[i]; }
#pragma unroll
    for (int t = 0; t < NT; t++) {
      v16bf b = *(const v16bf*)(sWr + (t * 16 + ln) * HD + k0 + 16 * h);
      c[t] = __builtin_amdgcn_wmma_f32_16x16x32_bf16(false, a, false, b,
                                                     (short)0, c[t], false, false);
    }
  }
  // bias
#pragma unroll
  for (int t = 0; t < NT; t++) {
    float bv = bias[t * 16 + ln];
#pragma unroll
    for (int r = 0; r < 8; r++) c[t][r] += bv;
  }
  if (NORMRELU) {
    // Row m = r + 8h spans the 16 lanes of this half across all tiles.
    float ss[8];
#pragma unroll
    for (int r = 0; r < 8; r++) {
      float s = 0.0f;
#pragma unroll
      for (int t = 0; t < NT; t++) s += c[t][r] * c[t][r];
      ss[r] = s;
    }
#pragma unroll
    for (int off = 1; off < 16; off <<= 1) {
#pragma unroll
      for (int r = 0; r < 8; r++) ss[r] += __shfl_xor(ss[r], off, 32);
    }
#pragma unroll
    for (int r = 0; r < 8; r++) {
      float s = 1.0f / fmaxf(sqrtf(ss[r]), 1e-12f);
#pragma unroll
      for (int t = 0; t < NT; t++) {
        float v = c[t][r] * s;
        c[t][r] = v > 0.0f ? v : 0.0f;
      }
    }
  }
  // store
  if (OUTBF16) {
    bf16* o = (bf16*)outv;
#pragma unroll
    for (int t = 0; t < NT; t++)
#pragma unroll
      for (int r = 0; r < 8; r++)
        o[((long)strip * 16 + r + 8 * h) * N + t * 16 + ln] = (bf16)c[t][r];
  } else {
    float* o = (float*)outv;
#pragma unroll
    for (int t = 0; t < NT; t++)
#pragma unroll
      for (int r = 0; r < 8; r++)
        o[((long)strip * 16 + r + 8 * h) * N + t * 16 + ln] = c[t][r];
  }
}

// ------------------------- supervision-edge dot -----------------------------
// one wave per edge: 64 channels -> 2 f32 per lane, xor-shuffle reduce.
__global__ __launch_bounds__(TPB) void dot_k(const int* __restrict__ iu,
                                             const int* __restrict__ ia,
                                             const float* __restrict__ u2,
                                             const float* __restrict__ a2,
                                             float* __restrict__ out, int EL) {
  int w = (int)(((long)blockIdx.x * blockDim.x + threadIdx.x) >> 5);
  if (w >= EL) return;
  int lane = threadIdx.x & 31;
  long ru = (long)iu[w] * OD, ra = (long)ia[w] * OD;
  float s = u2[ru + lane] * a2[ra + lane] + u2[ru + 32 + lane] * a2[ra + 32 + lane];
#pragma unroll
  for (int off = 16; off >= 1; off >>= 1) s += __shfl_xor(s, off, 32);
  if (lane == 0) out[w] = s;
}

// ----------------------------------------------------------------------------

extern "C" void kernel_launch(void* const* d_in, const int* in_sizes, int n_in,
                              void* d_out, int out_size, void* d_ws, size_t ws_size,
                              hipStream_t stream) {
  const int*   n_id_user = (const int*)  d_in[0];
  const float* x_user    = (const float*)d_in[1];
  const int*   n_id_app  = (const int*)  d_in[2];
  const float* x_app     = (const float*)d_in[3];
  const int*   edge_src  = (const int*)  d_in[4];
  const int*   edge_dst  = (const int*)  d_in[5];
  const int*   eli_user  = (const int*)  d_in[6];
  const int*   eli_app   = (const int*)  d_in[7];
  const float* emb_user  = (const float*)d_in[8];
  const float* emb_app   = (const float*)d_in[9];
  const float* Wlin_u = (const float*)d_in[10]; const float* blin_u = (const float*)d_in[11];
  const float* Wlin_a = (const float*)d_in[12]; const float* blin_a = (const float*)d_in[13];
  const float* W1u_l  = (const float*)d_in[14]; const float* b1u    = (const float*)d_in[15];
  const float* W1u_r  = (const float*)d_in[16];
  const float* W1a_l  = (const float*)d_in[17]; const float* b1a    = (const float*)d_in[18];
  const float* W1a_r  = (const float*)d_in[19];
  const float* W2u_l  = (const float*)d_in[20]; const float* b2u    = (const float*)d_in[21];
  const float* W2u_r  = (const float*)d_in[22];
  const float* W2a_l  = (const float*)d_in[23]; const float* b2a    = (const float*)d_in[24];
  const float* W2a_r  = (const float*)d_in[25];

  const int NU = in_sizes[0];
  const int NA = in_sizes[2];
  const int E  = in_sizes[4];
  const int EL = in_sizes[6];

  // ---- workspace carving (~322 MB) ----
  char* ws = (char*)d_ws;
  size_t off = 0;
  auto alloc = [&](size_t bytes) -> char* {
    char* p = ws + off;
    off = (off + bytes + 255) & ~(size_t)255;
    return p;
  };
  bf16*  hu     = (bf16*) alloc((size_t)NU * HD * 2);
  bf16*  ha     = (bf16*) alloc((size_t)NA * HD * 2);
  bf16*  u1     = (bf16*) alloc((size_t)NU * HD * 2);
  bf16*  a1     = (bf16*) alloc((size_t)NA * HD * 2);
  float* agg_u  = (float*)alloc((size_t)NU * HD * 4);
  float* agg_a  = (float*)alloc((size_t)NA * HD * 4);
  float* u2     = (float*)alloc((size_t)NU * OD * 4);
  float* a2     = (float*)alloc((size_t)NA * OD * 4);
  float* deg_u  = (float*)alloc((size_t)NU * 4);
  float* deg_a  = (float*)alloc((size_t)NA * 4);
  bf16* bWlin_u = (bf16*)alloc((size_t)HD * FUD * 2);
  bf16* bWlin_a = (bf16*)alloc((size_t)HD * FAD * 2);
  bf16* bW1u_l  = (bf16*)alloc((size_t)HD * HD * 2);
  bf16* bW1u_r  = (bf16*)alloc((size_t)HD * HD * 2);
  bf16* bW1a_l  = (bf16*)alloc((size_t)HD * HD * 2);
  bf16* bW1a_r  = (bf16*)alloc((size_t)HD * HD * 2);
  bf16* bW2u_l  = (bf16*)alloc((size_t)OD * HD * 2);
  bf16* bW2u_r  = (bf16*)alloc((size_t)OD * HD * 2);
  bf16* bW2a_l  = (bf16*)alloc((size_t)OD * HD * 2);
  bf16* bW2a_r  = (bf16*)alloc((size_t)OD * HD * 2);
  (void)ws_size; (void)n_in; (void)out_size;

  // zero accumulators + degrees
  zero_f32_k<<<2048, TPB, 0, stream>>>(agg_u, (long)NU * HD);
  zero_f32_k<<<1024, TPB, 0, stream>>>(agg_a, (long)NA * HD);
  zero_f32_k<<<256,  TPB, 0, stream>>>(deg_u, NU);
  zero_f32_k<<<64,   TPB, 0, stream>>>(deg_a, NA);

  // weight f32 -> bf16 conversion (tiny, runs once per launch)
  struct CvJob { const float* s; bf16* d; int n; };
  CvJob cv[10] = {
      {Wlin_u, bWlin_u, HD * FUD}, {Wlin_a, bWlin_a, HD * FAD},
      {W1u_l, bW1u_l, HD * HD},    {W1u_r, bW1u_r, HD * HD},
      {W1a_l, bW1a_l, HD * HD},    {W1a_r, bW1a_r, HD * HD},
      {W2u_l, bW2u_l, OD * HD},    {W2u_r, bW2u_r, OD * HD},
      {W2a_l, bW2a_l, OD * HD},    {W2a_r, bW2a_r, OD * HD}};
  for (int i = 0; i < 10; i++)
    cvt_bf16_k<<<(cv[i].n + TPB - 1) / TPB, TPB, 0, stream>>>(cv[i].s, cv[i].d, cv[i].n);

  // input encoders (WMMA)
  int strips_u = (NU + 15) / 16, strips_a = (NA + 15) / 16;
  int bu = (strips_u + 7) / 8, ba = (strips_a + 7) / 8;  // 8 waves / block
  encode_k<FUD><<<bu, TPB, 0, stream>>>(n_id_user, x_user, emb_user, bWlin_u, blin_u, hu, NU);
  encode_k<FAD><<<ba, TPB, 0, stream>>>(n_id_app,  x_app,  emb_app,  bWlin_a, blin_a, ha, NA);

  // degrees + layer-1 scatter
  deg_k<<<(E + TPB - 1) / TPB, TPB, 0, stream>>>(edge_src, edge_dst, deg_u, deg_a, E);
  long sct = (long)E * 32;
  int sblocks = (int)((sct + TPB - 1) / TPB);
  scatter_k<<<sblocks, TPB, 0, stream>>>(edge_src, edge_dst, hu, ha, agg_u, agg_a, E);

  // layer 1 (normalize + relu, bf16 out)
  sage_k<HD, true, true><<<bu, TPB, 0, stream>>>(agg_u, deg_u, hu, bW1u_l, bW1u_r, b1u, u1, NU);
  sage_k<HD, true, true><<<ba, TPB, 0, stream>>>(agg_a, deg_a, ha, bW1a_l, bW1a_r, b1a, a1, NA);

  // re-zero aggs, layer-2 scatter
  zero_f32_k<<<2048, TPB, 0, stream>>>(agg_u, (long)NU * HD);
  zero_f32_k<<<1024, TPB, 0, stream>>>(agg_a, (long)NA * HD);
  scatter_k<<<sblocks, TPB, 0, stream>>>(edge_src, edge_dst, u1, a1, agg_u, agg_a, E);

  // layer 2 (no norm, f32 out, N=64)
  sage_k<OD, false, false><<<bu, TPB, 0, stream>>>(agg_u, deg_u, u1, bW2u_l, bW2u_r, b2u, u2, NU);
  sage_k<OD, false, false><<<ba, TPB, 0, stream>>>(agg_a, deg_a, a1, bW2a_l, bW2a_r, b2a, a2, NA);

  // classifier: dot over supervision edges
  dot_k<<<(EL + 7) / 8, TPB, 0, stream>>>(eli_user, eli_app, u2, a2, (float*)d_out, EL);
}
